// EnhancedVisualOdometryModel_9363028706364
// MI455X (gfx1250) — matrix operand order repaired and checked by reading, not compiled, not run on hardware
//
#include <hip/hip_runtime.h>
#include <hip/hip_bf16.h>

// ---------------------------------------------------------------------------
// CDNA5 (gfx1250) implementation of EnhancedVisualOdometryModel forward.
// Conv + linear layers: implicit GEMM on v_wmma_f32_16x16x32_f16.
// Block = 128 threads (4 waves), tile M=64 x N=64, K chunks of 64:
//   - threads 0..63  stage the 64x64 im2col B tile (one column each,
//     incremental k->(ci,fy,fx) decomposition: no divisions in the hot loop)
//   - threads 64..127 stage the 64x64 weight A tile (one row each, linear)
//   - tiles land in LDS pre-swizzled into WMMA fragment order (ds_store_b32),
//     each wave consumes them as contiguous v16h loads (ds_load_b128)
//   - 8 v_wmma per wave per chunk; fused bias / eval-BN / activation epilogue
// ---------------------------------------------------------------------------

typedef __attribute__((ext_vector_type(16))) _Float16 v16h;
typedef __attribute__((ext_vector_type(2)))  _Float16 v2h;
typedef __attribute__((ext_vector_type(8)))  float    v8f;

#define ACT_NONE     0
#define ACT_RELU     1
#define ACT_SIGMOID  2
#define ACT_SOFTPLUS 3

struct ConvParams {
  const float* in;     // [B, Cin, Hi, Wi] (batch stride overridable)
  const float* wgt;    // [Cout, Cin, kh, kw] row-major == [Cout, K]
  const float* bias;   // [Cout] or null
  const float* gamma;  // BN gamma or null
  const float* beta;   // BN beta (valid iff gamma)
  float* out;          // [B, Cout, Ho, Wo]
  int B, Cin, Hi, Wi, Cout, Ho, Wo, kh, kw, stride, pad;
  long inBatchStride;  // elements between batches of `in`
  int act;
};

__global__ __launch_bounds__(128) void conv_wmma(ConvParams p) {
  // Fragment-ordered tiles: [wave|ntile][K-subtile][32 lanes][16 halves]
  __shared__ __align__(32) _Float16 lA[4096];   // 4 waves  x 2 x 512
  __shared__ __align__(32) _Float16 lB[4096];   // 4 ntiles x 2 x 512

  const int tid  = threadIdx.x;
  const int lane = tid & 31;
  const int wave = tid >> 5;
  const int HoWo = p.Ho * p.Wo;
  const int HiWi = p.Hi * p.Wi;
  const int N = p.B * HoWo;
  const int K = p.Cin * p.kh * p.kw;
  const int mBlocks = (p.Cout + 63) >> 6;
  const int mBase = (blockIdx.x % mBlocks) << 6;
  const int nBase = (blockIdx.x / mBlocks) << 6;

  // ---- B-staging thread state (threads 0..63): one column, incremental k.
  const int nT = nBase + tid;                   // column owned (tid < 64)
  const bool nTOK = (tid < 64) && (nT < N);
  int bT = 0, hoT = 0, woT = 0;
  if (nTOK) { bT = nT / HoWo; int r = nT - bT * HoWo; hoT = r / p.Wo; woT = r - hoT * p.Wo; }
  const float* inT = p.in + (size_t)bT * (size_t)p.inBatchStride;
  int ci = 0, fy = 0, fx = 0;                   // k -> (ci,fy,fx), carried

  // ---- A-staging thread state (threads 64..127): one weight row, linear k.
  const int at = tid - 64;
  const int mT = mBase + at;
  const bool mTOK = (tid >= 64) && (mT < p.Cout);
  const float* wrow = p.wgt + (size_t)(mTOK ? mT : 0) * (size_t)K;

  v8f acc[4] = {};
  for (int k0 = 0; k0 < K; k0 += 64) {
    __syncthreads();                            // previous-iteration readers done

    if (tid < 64) {
      // ---- stage B: fragment slot for k_local kl: hs=(kl&31)>>4, j=kl&15,
      //      bl=(col&15)+16*hs, region = (col>>4)*1024 + (kl>>5)*512
      const int colReg = (tid >> 4) << 10;
      const int colLn  = tid & 15;
#pragma unroll
      for (int kk = 0; kk < 64; kk += 2) {
        float f[2];
#pragma unroll
        for (int u = 0; u < 2; ++u) {
          const int k = k0 + kk + u;
          float val = 0.0f;
          if (nTOK && k < K) {
            const int hi = hoT * p.stride - p.pad + fy;
            const int wi = woT * p.stride - p.pad + fx;
            if ((unsigned)hi < (unsigned)p.Hi && (unsigned)wi < (unsigned)p.Wi)
              val = inT[(size_t)ci * HiWi + (size_t)hi * p.Wi + wi];
          }
          f[u] = val;
          if (++fx == p.kw) { fx = 0; if (++fy == p.kh) { fy = 0; ++ci; } }
        }
        const int sub = kk >> 5, r = kk & 31;
        const int hsB = r >> 4, j = r & 15;     // j even -> 4B-aligned pair
        v2h pr; pr[0] = (_Float16)f[0]; pr[1] = (_Float16)f[1];
        *(v2h*)&lB[colReg + (sub << 9) + (((colLn + (hsB << 4)) << 4) + j)] = pr;
      }
    } else {
      // ---- stage A: fragment slot for k_local kl: hs=(kl&31)>>4,
      //      j = kl&15, al = (row&15)+16*hs
      const int wReg = (at >> 4) << 10;
      const int rowLn = at & 15;
#pragma unroll
      for (int kk = 0; kk < 64; kk += 2) {
        const int ka = k0 + kk;
        const float f0 = (mTOK && ka     < K) ? wrow[ka]     : 0.0f;
        const float f1 = (mTOK && ka + 1 < K) ? wrow[ka + 1] : 0.0f;
        const int sub = kk >> 5, r = kk & 31, g = r >> 3;
        const int hsA = r >> 4;
        const int j = r & 15;
        v2h pr; pr[0] = (_Float16)f0; pr[1] = (_Float16)f1;
        *(v2h*)&lA[wReg + (sub << 9) + (((rowLn + (hsA << 4)) << 4) + j)] = pr;
      }
      // Prefetch next K-chunk of this row (global_prefetch_b8).
      if (k0 + 64 < K) __builtin_prefetch(&wrow[k0 + 64], 0, 1);
    }

    __syncthreads();                            // tiles visible to all waves

    // ---- consume: wave owns A rows [mBase+16*wave, +16), sweeps 4 N-subtiles
    const v16h a0 = *(const v16h*)&lA[(wave << 10) + lane * 16];
    const v16h a1 = *(const v16h*)&lA[(wave << 10) + 512 + lane * 16];
#pragma unroll
    for (int nt = 0; nt < 4; ++nt) {
      const v16h b0 = *(const v16h*)&lB[(nt << 10) + lane * 16];
      const v16h b1 = *(const v16h*)&lB[(nt << 10) + 512 + lane * 16];
      acc[nt] = __builtin_amdgcn_wmma_f32_16x16x32_f16(
          false, a0, false, b0, (short)0, acc[nt], false, false);
      acc[nt] = __builtin_amdgcn_wmma_f32_16x16x32_f16(
          false, a1, false, b1, (short)0, acc[nt], false, false);
    }
  }

  // ---- fused epilogue: D layout VGPR r -> M = r (+8 for lanes 16..31)
  const float bnscale = 0.9999950000374997f;    // rsqrt(1 + 1e-5)
  const int hs  = lane >> 4;
  const int l15 = lane & 15;
#pragma unroll
  for (int nt = 0; nt < 4; ++nt) {
    const int n = nBase + (nt << 4) + l15;
    if (n < N) {
      const int nb = n / HoWo;
      const int rr = n - nb * HoWo;
#pragma unroll
      for (int r = 0; r < 8; ++r) {
        const int mo = mBase + (wave << 4) + r + (hs << 3);
        if (mo < p.Cout) {
          float v = acc[nt][r];
          if (p.bias)  v += p.bias[mo];
          if (p.gamma) v = v * (p.gamma[mo] * bnscale) + p.beta[mo];
          if (p.act == ACT_RELU)          v = fmaxf(v, 0.0f);
          else if (p.act == ACT_SIGMOID)  v = 1.0f / (1.0f + __expf(-v));
          else if (p.act == ACT_SOFTPLUS) v = (v > 20.0f) ? v : log1pf(__expf(v));
          p.out[((size_t)nb * p.Cout + mo) * (size_t)HoWo + (size_t)rr] = v;
        }
      }
    }
  }
}

// ---------------------------------------------------------------------------
// Pointwise / reduction helpers
// ---------------------------------------------------------------------------

__global__ void maxpool3x3s2_kernel(const float* in, float* out,
                                    int BC, int Hi, int Wi, int Ho, int Wo) {
  int idx = blockIdx.x * blockDim.x + threadIdx.x;
  if (idx >= BC * Ho * Wo) return;
  int wo = idx % Wo; int t = idx / Wo;
  int ho = t % Ho;   t /= Ho;
  const float* base = in + (size_t)t * Hi * Wi;
  float m = -3.402823466e38f;
  for (int dy = -1; dy <= 1; ++dy) {
    int hi = ho * 2 + dy;
    if ((unsigned)hi >= (unsigned)Hi) continue;
    for (int dx = -1; dx <= 1; ++dx) {
      int wi = wo * 2 + dx;
      if ((unsigned)wi >= (unsigned)Wi) continue;
      m = fmaxf(m, base[hi * Wi + wi]);
    }
  }
  out[idx] = m;
}

__global__ void add_relu_kernel(const float* a, const float* b, float* out, int n) {
  int i = blockIdx.x * blockDim.x + threadIdx.x;
  if (i < n) out[i] = fmaxf(a[i] + b[i], 0.0f);
}

__global__ void mul_kernel(float* a, const float* b, int n) {
  int i = blockIdx.x * blockDim.x + threadIdx.x;
  if (i < n) a[i] *= b[i];
}

// out[b*outStride + outOff + c] = mean over HW of in[b,c,:,:]
__global__ void avgpool_kernel(const float* in, float* out, int B, int C, int HW,
                               int outStride, int outOff) {
  int idx = blockIdx.x * blockDim.x + threadIdx.x;
  if (idx >= B * C) return;
  int b = idx / C, c = idx % C;
  const float* p = in + ((size_t)b * C + c) * (size_t)HW;
  float s = 0.0f;
  for (int i = 0; i < HW; ++i) s += p[i];
  out[(size_t)b * outStride + outOff + c] = s / (float)HW;
}

__global__ void copy_strided_kernel(const float* src, int srcStride,
                                    float* dst, int dstStride, int dstOff,
                                    int rows, int cols) {
  int idx = blockIdx.x * blockDim.x + threadIdx.x;
  if (idx >= rows * cols) return;
  int r = idx / cols, c = idx % cols;
  dst[(size_t)r * dstStride + dstOff + c] = src[(size_t)r * srcStride + c];
}

// 13x13 shift correlation minus center: out[b, o, h, w], o in [0,168)
__global__ void correlation_kernel(const float* x1, const float* x2, float* out,
                                   int B, int C, int H, int W) {
  int idx = blockIdx.x * blockDim.x + threadIdx.x;
  const int OC = 168;
  int total = B * OC * H * W;
  if (idx >= total) return;
  int w = idx % W; int t = idx / W;
  int h = t % H;   t /= H;
  int o = t % OC;  int b = t / OC;
  int op = (o < 84) ? o : o + 1;        // skip (0,0) at grid index 84
  int di = op / 13 - 6, dj = op % 13 - 6;
  int h2 = h + di, w2 = w + dj;
  float s = 0.0f;
  if ((unsigned)h2 < (unsigned)H && (unsigned)w2 < (unsigned)W) {
    const size_t HW = (size_t)H * W;
    const float* p1 = x1 + (size_t)b * C * HW + (size_t)h * W + w;
    const float* p2 = x2 + (size_t)b * C * HW + (size_t)h2 * W + w2;
    for (int c = 0; c < C; ++c) s += p1[c * HW] * p2[c * HW];
  }
  out[idx] = s * rsqrtf((float)C);
}

// rot = (1,0,0,0); trans = fc3 * scale. d_out = [rot(B*4) | trans(B*3)]
__global__ void finalize_kernel(const float* t, const float* scl, float* out, int B) {
  int i = threadIdx.x;
  if (i < B * 4) out[i] = ((i & 3) == 0) ? 1.0f : 0.0f;
  int j = i - B * 4;
  if (j >= 0 && j < B * 3) out[B * 4 + j] = t[j] * scl[j / 3];
}

// ---------------------------------------------------------------------------
// Host orchestration
// ---------------------------------------------------------------------------

extern "C" void kernel_launch(void* const* d_in, const int* in_sizes, int n_in,
                              void* d_out, int out_size, void* d_ws, size_t ws_size,
                              hipStream_t stream) {
  const int B = 8;
  auto F = [&](int i) { return (const float*)d_in[i]; };

  // jax tree_flatten (sorted dict keys) leaf order for setup_inputs().
  enum {
    IMG1 = 0, IMG2 = 1,
    ATT_B = 2, ATT_W = 3, BN1_B = 4, BN1_G = 5,
    CC1_B = 6, CC1_BB = 7, CC1_G = 8, CC1_W = 9,
    CC2_B = 10, CC2_BB = 11, CC2_G = 12, CC2_W = 13,
    CC3_B = 14, CC3_BB = 15, CC3_G = 16, CC3_W = 17,
    CONV1_W = 18,
    FC1_B = 19, FC1_BB = 20, FC1_G = 21, FC1_W = 22,
    FC2_B = 23, FC2_BB = 24, FC2_G = 25, FC2_W = 26,
    FC3_B = 27, FC3_W = 28,
    GC1_B = 29, GC1_W = 30, GC2_B = 31, GC2_W = 32,
    L1B0 = 33,   // no-ds block: b1,b2,c1,c2,g1,g2
    L1B1 = 39,
    L2B0 = 45,   // ds block: b1,b2,bd,c1,c2,cd,g1,g2,gd
    L2B1 = 54,
    L3B0 = 60,
    L3B1 = 69,
    REF1_B = 75, REF1_BB = 76, REF1_G = 77, REF1_W = 78,
    REF2_B = 79, REF2_BB = 80, REF2_G = 81, REF2_W = 82,
    SB1_B = 83, SB1_BB = 84, SB1_G = 85, SB1_W = 86,
    SB2_B = 87, SB2_BB = 88, SB2_G = 89, SB2_W = 90,
    SB3_B = 91, SB3_W = 92
  };

  auto conv = [&](const float* in, const float* w, const float* bias,
                  const float* g, const float* bb, float* out,
                  int Cin, int Hi, int Wi, int Cout, int kh, int kw,
                  int stride, int pad, int act, long inBS) {
    ConvParams p;
    p.in = in; p.wgt = w; p.bias = bias; p.gamma = g; p.beta = bb; p.out = out;
    p.B = B; p.Cin = Cin; p.Hi = Hi; p.Wi = Wi; p.Cout = Cout;
    p.Ho = (Hi + 2 * pad - kh) / stride + 1;
    p.Wo = (Wi + 2 * pad - kw) / stride + 1;
    p.kh = kh; p.kw = kw; p.stride = stride; p.pad = pad;
    p.inBatchStride = inBS ? inBS : (long)Cin * Hi * Wi;
    p.act = act;
    long N = (long)B * p.Ho * p.Wo;
    long mBlocks = (Cout + 63) / 64;
    long nBlocks = (N + 63) / 64;
    conv_wmma<<<dim3((unsigned)(mBlocks * nBlocks)), 128, 0, stream>>>(p);
  };

  // ---- workspace carve (floats) -------------------------------------------
  float* ws = (float*)d_ws;
  size_t off = 0;
  auto alloc = [&](size_t n) { float* r = ws + off; off += n; return r; };
  float* low0 = alloc(2097152);   // [8,128,32,64]
  float* low1 = alloc(2097152);
  float* comb = alloc(8 * 448);   // [corr_feat | r1 | r2 | gc1]
  float* gcv  = alloc(8 * 256);
  float* g1v  = alloc(8 * 128);
  float* g2v  = alloc(8 * 128);
  float* tf1  = alloc(8 * 512);
  float* tf2  = alloc(8 * 256);
  float* fc3o = alloc(32);
  float* s1v  = alloc(8 * 128);
  float* s2v  = alloc(8 * 64);
  float* sclv = alloc(32);
  float* temp = alloc(20971520);  // 84 MB reusable arena
  float* T1 = temp;
  float* T2 = temp + 4194304;
  float* T3 = temp + 8388608;
  float* T4 = temp + 12582912;    // ping-pong B
  float* Xb = temp + 16777216;    // ping-pong A (pool output)
  float* c1out = temp;            // [8,64,128,256] — dead before T1..T4 go live

  // ---- feature extraction for both images ---------------------------------
  for (int im = 0; im < 2; ++im) {
    const float* img = F(im == 0 ? IMG1 : IMG2);
    conv(img, F(CONV1_W), nullptr, F(BN1_G), F(BN1_B), c1out,
         3, 256, 512, 64, 7, 7, 2, 3, ACT_RELU, 0);
    {
      int n = B * 64 * 64 * 128;
      maxpool3x3s2_kernel<<<(n + 255) / 256, 256, 0, stream>>>(c1out, Xb, B * 64, 128, 256, 64, 128);
    }
    float* cur = Xb;
    float* nxt = T4;
    auto block = [&](int base, bool ds, int Cin, int Cout, int Hi, int Wi, int stride) {
      const float* b1 = F(base + 0);
      const float* b2 = F(base + 1);
      const float* c1 = F(base + (ds ? 3 : 2));
      const float* c2 = F(base + (ds ? 4 : 3));
      const float* g1 = F(base + (ds ? 6 : 4));
      const float* g2 = F(base + (ds ? 7 : 5));
      int Ho = (Hi + 2 - 3) / stride + 1;
      int Wo = (Wi + 2 - 3) / stride + 1;
      conv(cur, c1, nullptr, g1, b1, T1, Cin, Hi, Wi, Cout, 3, 3, stride, 1, ACT_RELU, 0);
      conv(T1, c2, nullptr, g2, b2, T2, Cout, Ho, Wo, Cout, 3, 3, 1, 1, ACT_NONE, 0);
      const float* idn = cur;
      if (ds) {
        conv(cur, F(base + 5), nullptr, F(base + 8), F(base + 2), T3,
             Cin, Hi, Wi, Cout, 1, 1, stride, 0, ACT_NONE, 0);
        idn = T3;
      }
      int nelem = B * Cout * Ho * Wo;
      add_relu_kernel<<<(nelem + 255) / 256, 256, 0, stream>>>(T2, idn, nxt, nelem);
      float* t = cur; cur = nxt; nxt = t;
    };
    block(L1B0, false, 64, 64, 64, 128, 1);
    block(L1B1, false, 64, 64, 64, 128, 1);
    block(L2B0, true, 64, 128, 64, 128, 2);
    block(L2B1, false, 128, 128, 32, 64, 1);
    hipMemcpyAsync(im == 0 ? low0 : low1, cur, (size_t)2097152 * sizeof(float),
                   hipMemcpyDeviceToDevice, stream);
    block(L3B0, true, 128, 256, 32, 64, 2);
    block(L3B1, false, 256, 256, 16, 32, 1);
    // cur == high [8,256,16,32]
    conv(cur, F(REF1_W), F(REF1_B), F(REF1_G), F(REF1_BB), T1,
         256, 16, 32, 256, 3, 3, 1, 1, ACT_RELU, 0);
    conv(T1, F(REF2_W), F(REF2_B), F(REF2_G), F(REF2_BB), T2,
         256, 16, 32, 128, 3, 3, 1, 1, ACT_RELU, 0);
    avgpool_kernel<<<(8 * 128 + 255) / 256, 256, 0, stream>>>(T2, comb, B, 128, 512,
                                                              448, 64 + im * 128);
    if (im == 0) {
      avgpool_kernel<<<(8 * 256 + 255) / 256, 256, 0, stream>>>(cur, gcv, B, 256, 512, 256, 0);
      conv(gcv, F(GC1_W), F(GC1_B), nullptr, nullptr, g1v,
           256, 1, 1, 128, 1, 1, 1, 0, ACT_RELU, 0);
      conv(g1v, F(GC2_W), F(GC2_B), nullptr, nullptr, g2v,
           128, 1, 1, 128, 1, 1, 1, 0, ACT_SIGMOID, 0);
      copy_strided_kernel<<<(8 * 128 + 255) / 256, 256, 0, stream>>>(g2v, 128, comb, 448, 320, 8, 128);
    }
  }

  // ---- correlation head ----------------------------------------------------
  float* corr = temp;
  float* att  = temp + 2752512;
  float* cc1o = temp + 5505024;
  float* cc2o = temp + 9699328;
  float* cc3o = temp + 11796480;
  int ncorr = B * 168 * 32 * 64;
  correlation_kernel<<<(ncorr + 255) / 256, 256, 0, stream>>>(low0, low1, corr, B, 128, 32, 64);
  conv(corr, F(ATT_W), F(ATT_B), nullptr, nullptr, att,
       168, 32, 64, 168, 1, 1, 1, 0, ACT_SIGMOID, 0);
  mul_kernel<<<(ncorr + 255) / 256, 256, 0, stream>>>(corr, att, ncorr);
  conv(corr, F(CC1_W), F(CC1_B), F(CC1_G), F(CC1_BB), cc1o, 168, 32, 64, 256, 3, 3, 1, 1, ACT_RELU, 0);
  conv(cc1o, F(CC2_W), F(CC2_B), F(CC2_G), F(CC2_BB), cc2o, 256, 32, 64, 128, 3, 3, 1, 1, ACT_RELU, 0);
  conv(cc2o, F(CC3_W), F(CC3_B), F(CC3_G), F(CC3_BB), cc3o, 128, 32, 64, 64, 3, 3, 1, 1, ACT_RELU, 0);
  avgpool_kernel<<<(8 * 64 + 255) / 256, 256, 0, stream>>>(cc3o, comb, B, 64, 2048, 448, 0);

  // ---- FC heads (linear == 1x1 conv on [B,C,1,1], row stride overridable) --
  conv(comb,      F(FC1_W), F(FC1_B), F(FC1_G), F(FC1_BB), tf1, 448, 1, 1, 512, 1, 1, 1, 0, ACT_RELU, 448);
  conv(tf1,       F(FC2_W), F(FC2_B), F(FC2_G), F(FC2_BB), tf2, 512, 1, 1, 256, 1, 1, 1, 0, ACT_RELU, 0);
  conv(tf2,       F(FC3_W), F(FC3_B), nullptr, nullptr,    fc3o, 256, 1, 1, 3, 1, 1, 1, 0, ACT_NONE, 0);
  conv(comb + 64, F(SB1_W), F(SB1_B), F(SB1_G), F(SB1_BB), s1v, 256, 1, 1, 128, 1, 1, 1, 0, ACT_RELU, 448);
  conv(s1v,       F(SB2_W), F(SB2_B), F(SB2_G), F(SB2_BB), s2v, 128, 1, 1, 64, 1, 1, 1, 0, ACT_RELU, 0);
  conv(s2v,       F(SB3_W), F(SB3_B), nullptr, nullptr,    sclv, 64, 1, 1, 1, 1, 1, 1, 0, ACT_SOFTPLUS, 0);

  finalize_kernel<<<1, 64, 0, stream>>>(fc3o, sclv, (float*)d_out, B);
}
